// MultiHeadSelfAttention_38457137169075
// MI455X (gfx1250) — matrix operand order repaired and checked by reading, not compiled, not run on hardware
//
#include <hip/hip_runtime.h>
#include <stdint.h>

#define Bn   4
#define Tn   2048
#define Dn   1024
#define Hn   16
#define HDn  64
#define NQKV 3072
#define MR   (Bn*Tn)   // 8192 token rows

typedef __attribute__((ext_vector_type(16))) __bf16 v16bf;
typedef __attribute__((ext_vector_type(8)))  float  v8f;

union Frag {
  v16bf v;
  unsigned int   u[8];
  unsigned short us[16];
};

__device__ inline unsigned short f2bf(float f) {
  union { float f; unsigned u; } x; x.f = f;
  unsigned u = x.u;
  u += 0x7FFFu + ((u >> 16) & 1u);           // round-to-nearest-even
  return (unsigned short)(u >> 16);
}

__device__ inline v8f vzero() {
  v8f z;
#pragma unroll
  for (int i = 0; i < 8; ++i) z[i] = 0.0f;
  return z;
}

// A-matrix (16x32 bf16) per-lane K offset for packed pair j (contiguous: K = 16*hh + 2*j)
__device__ inline int koffA(int j, int hh) { return 16 * hh + 2 * j; }

// ---------------------------------------------------------------------------
// Prep kernels: one-time fp32 -> bf16 conversions (hoisted out of GEMM loops)
// ---------------------------------------------------------------------------
__global__ __launch_bounds__(256) void cvt_x_kernel(
    const float* __restrict__ src, unsigned short* __restrict__ dst, int n4) {
  const int t = blockIdx.x * blockDim.x + threadIdx.x;
  if (t >= n4) return;
  float4 f = ((const float4*)src)[t];
  uint2 o;
  o.x = (unsigned)f2bf(f.x) | ((unsigned)f2bf(f.y) << 16);
  o.y = (unsigned)f2bf(f.z) | ((unsigned)f2bf(f.w) << 16);
  ((uint2*)dst)[t] = o;
}

// src [K][N] fp32 row-major  ->  dst [N][K] bf16 (K contiguous)
__global__ __launch_bounds__(256) void cvt_t_kernel(
    const float* __restrict__ src, unsigned short* __restrict__ dst, int K, int N) {
  const int t = blockIdx.x * blockDim.x + threadIdx.x;
  if (t >= K * N) return;
  const int k = t / N, n = t - k * N;
  dst[(size_t)n * K + k] = f2bf(src[t]);
}

// ---------------------------------------------------------------------------
// Kernel 1: qkv = x @ W_qkv, all-bf16 operands. Wave tile 64(M) x 128(N).
// Scatter: Q,K -> [B,H,T,64];  V -> [B,H,64,T] (pre-transposed for PV stage).
// ---------------------------------------------------------------------------
__global__ __launch_bounds__(128) void qkv_gemm_kernel(
    const unsigned short* __restrict__ Xb, const unsigned short* __restrict__ Wt,
    unsigned short* __restrict__ Qb, unsigned short* __restrict__ Kb,
    unsigned short* __restrict__ Vt) {
  const int lane = threadIdx.x & 31;
  const int wid  = (blockIdx.x * blockDim.x + threadIdx.x) >> 5;
  const int NT   = NQKV / 128;                // 24 tiles along N
  const int m0   = (wid / NT) * 64;
  const int n0   = (wid % NT) * 128;
  const int ln   = lane & 15, hh = lane >> 4;

  v8f acc[4][8];
#pragma unroll
  for (int i = 0; i < 4; ++i)
#pragma unroll
    for (int j = 0; j < 8; ++j) acc[i][j] = vzero();

  for (int kc = 0; kc < Dn / 32; ++kc) {
    const int kbase = kc * 32;
    Frag a[4];
#pragma unroll
    for (int ms = 0; ms < 4; ++ms) {
      const unsigned short* ap = Xb + (size_t)(m0 + ms * 16 + ln) * Dn + kbase;
      __builtin_prefetch(ap + 128, 0, 0);     // global_prefetch_b8, next chunks
#pragma unroll
      for (int j = 0; j < 8; ++j)
        a[ms].u[j] = *(const unsigned int*)(ap + koffA(j, hh));
    }
#pragma unroll
    for (int ns = 0; ns < 8; ++ns) {
      Frag bf;                                 // B 32x16: lane = N col, K = 16*hh + e
      const unsigned short* bp =
          Wt + (size_t)(n0 + ns * 16 + ln) * Dn + kbase + hh * 16;
      __builtin_prefetch(bp + 128, 0, 0);
#pragma unroll
      for (int j = 0; j < 8; ++j) bf.u[j] = *(const unsigned int*)(bp + 2 * j);
#pragma unroll
      for (int ms = 0; ms < 4; ++ms)
        acc[ms][ns] = __builtin_amdgcn_wmma_f32_16x16x32_bf16(
            false, a[ms].v, false, bf.v, (short)0, acc[ms][ns], false, false);
    }
  }

#pragma unroll
  for (int ms = 0; ms < 4; ++ms)
#pragma unroll
    for (int ns = 0; ns < 8; ++ns)
#pragma unroll
      for (int r = 0; r < 8; ++r) {
        const int row  = m0 + ms * 16 + r + 8 * hh;   // token row
        const int col  = n0 + ns * 16 + ln;           // [0,3072)
        const int part = col >> 10;                   // 0=Q 1=K 2=V
        const int d    = col & 1023;
        const int head = d >> 6, hd = d & 63;
        const int b    = row >> 11, t = row & (Tn - 1);
        const unsigned short val = f2bf(acc[ms][ns][r]);
        const int bh = b * Hn + head;
        if (part == 0)      Qb[((size_t)bh * Tn + t) * HDn + hd] = val;
        else if (part == 1) Kb[((size_t)bh * Tn + t) * HDn + hd] = val;
        else                Vt[((size_t)bh * HDn + hd) * Tn + t] = val;
      }
}

// ---------------------------------------------------------------------------
// Kernel 2: flash attention per (b,h). One wave = 32 query rows (2 M-tiles),
// keys streamed in blocks of 32. K/V fragments shared across both M-tiles.
// ---------------------------------------------------------------------------
__global__ __launch_bounds__(128) void attn_kernel(
    const unsigned short* __restrict__ Qb, const unsigned short* __restrict__ Kb,
    const unsigned short* __restrict__ Vt, const float* __restrict__ Wrel,
    unsigned short* __restrict__ Ob) {
  __shared__ unsigned short lds_p[4][32 * 32];      // per-wave P tile (bf16)
  const int lane  = threadIdx.x & 31;
  const int wslot = threadIdx.x >> 5;
  const int wid   = (blockIdx.x * blockDim.x + threadIdx.x) >> 5;
  const int nqb   = Tn / 32;                        // 64 query blocks per (b,h)
  const int bh    = wid / nqb;
  const int q0    = (wid % nqb) * 32;
  const int ln    = lane & 15, hh = lane >> 4;
  const int h     = bh % Hn, b = bh / Hn;
  const float wr  = Wrel[h];

  const unsigned short* Qh = Qb + (size_t)bh * Tn * HDn;
  const unsigned short* Kh = Kb + (size_t)bh * Tn * HDn;
  const unsigned short* Vh = Vt + (size_t)bh * HDn * Tn;   // [64][T]

  // Q fragments: 2 M-tiles x 2 K-chunks, loaded once
  Frag qf[2][2];
#pragma unroll
  for (int ms = 0; ms < 2; ++ms)
#pragma unroll
    for (int kc = 0; kc < 2; ++kc) {
      const unsigned short* qp = Qh + (size_t)(q0 + ms * 16 + ln) * HDn + kc * 32;
#pragma unroll
      for (int j = 0; j < 8; ++j)
        qf[ms][kc].u[j] = *(const unsigned int*)(qp + koffA(j, hh));
    }

  float mrow[2][8], lrow[2][8];
#pragma unroll
  for (int ms = 0; ms < 2; ++ms)
#pragma unroll
    for (int r = 0; r < 8; ++r) { mrow[ms][r] = -__builtin_inff(); lrow[ms][r] = 0.0f; }
  v8f oacc[2][4];
#pragma unroll
  for (int ms = 0; ms < 2; ++ms)
#pragma unroll
    for (int ns = 0; ns < 4; ++ns) oacc[ms][ns] = vzero();

  for (int kb = 0; kb < Tn; kb += 32) {
    // ---- S = Q K^T (2 M-tiles x 2 key tiles), K frags shared across M
    v8f s[2][2];
#pragma unroll
    for (int ms = 0; ms < 2; ++ms) { s[ms][0] = vzero(); s[ms][1] = vzero(); }
#pragma unroll
    for (int kc = 0; kc < 2; ++kc)
#pragma unroll
      for (int ns = 0; ns < 2; ++ns) {
        Frag kf;  // B layout: lane = key col, elements = contiguous hd
        const unsigned short* kp =
            Kh + (size_t)(kb + ns * 16 + ln) * HDn + kc * 32 + hh * 16;
#pragma unroll
        for (int j = 0; j < 8; ++j) kf.u[j] = *(const unsigned int*)(kp + 2 * j);
#pragma unroll
        for (int ms = 0; ms < 2; ++ms)
          s[ms][ns] = __builtin_amdgcn_wmma_f32_16x16x32_bf16(
              false, qf[ms][kc].v, false, kf.v, (short)0, s[ms][ns], false, false);
      }

    // ---- online softmax, rows M = ms*16 + r + 8*hh
#pragma unroll
    for (int ms = 0; ms < 2; ++ms)
#pragma unroll
      for (int r = 0; r < 8; ++r) {
        const float qi = (float)(q0 + ms * 16 + r + 8 * hh);
        const float v0 = s[ms][0][r] * 0.125f + ((float)(kb + ln)      - qi) * wr;
        const float v1 = s[ms][1][r] * 0.125f + ((float)(kb + 16 + ln) - qi) * wr;
        float mx = fmaxf(v0, v1);
#pragma unroll
        for (int off = 1; off < 16; off <<= 1) mx = fmaxf(mx, __shfl_xor(mx, off, 32));
        const float mnew  = fmaxf(mrow[ms][r], mx);
        const float alpha = __expf(mrow[ms][r] - mnew);
        const float e0 = __expf(v0 - mnew), e1 = __expf(v1 - mnew);
        float ps = e0 + e1;
#pragma unroll
        for (int off = 1; off < 16; off <<= 1) ps += __shfl_xor(ps, off, 32);
        lrow[ms][r] = lrow[ms][r] * alpha + ps;
        mrow[ms][r] = mnew;
#pragma unroll
        for (int ns = 0; ns < 4; ++ns) oacc[ms][ns][r] *= alpha;
        const int prow = ms * 16 + r + 8 * hh;
        lds_p[wslot][prow * 32 + ln]      = f2bf(e0);
        lds_p[wslot][prow * 32 + 16 + ln] = f2bf(e1);
      }

    // ---- O += P @ V   (P via LDS transpose; V frags from [64][T] layout)
    Frag pf[2];
#pragma unroll
    for (int ms = 0; ms < 2; ++ms)
#pragma unroll
      for (int j = 0; j < 8; ++j)
        pf[ms].u[j] =
            *(const unsigned int*)&lds_p[wslot][(ms * 16 + ln) * 32 + koffA(j, hh)];
#pragma unroll
    for (int ns = 0; ns < 4; ++ns) {
      Frag vf;  // B layout: lane = hd col, elements = contiguous keys
      const unsigned short* vp = Vh + (size_t)(ns * 16 + ln) * Tn + kb + hh * 16;
#pragma unroll
      for (int j = 0; j < 8; ++j) vf.u[j] = *(const unsigned int*)(vp + 2 * j);
#pragma unroll
      for (int ms = 0; ms < 2; ++ms)
        oacc[ms][ns] = __builtin_amdgcn_wmma_f32_16x16x32_bf16(
            false, pf[ms].v, false, vf.v, (short)0, oacc[ms][ns], false, false);
    }
  }

  // normalize, store to [B,T,D] bf16 (head h occupies cols h*64..)
#pragma unroll
  for (int ms = 0; ms < 2; ++ms)
#pragma unroll
    for (int r = 0; r < 8; ++r) {
      const float inv = 1.0f / lrow[ms][r];
      const int   t   = q0 + ms * 16 + r + 8 * hh;
      const size_t base = ((size_t)(b * Tn + t)) * Dn + h * HDn;
#pragma unroll
      for (int ns = 0; ns < 4; ++ns)
        Ob[base + ns * 16 + ln] = f2bf(oacc[ms][ns][r] * inv);
    }
}

// ---------------------------------------------------------------------------
// Kernel 3: out = O @ W_proj, bf16 operands, fp32 out. Wave tile 64x128.
// ---------------------------------------------------------------------------
__global__ __launch_bounds__(128) void proj_gemm_kernel(
    const unsigned short* __restrict__ Ob, const unsigned short* __restrict__ Wt,
    float* __restrict__ out) {
  const int lane = threadIdx.x & 31;
  const int wid  = (blockIdx.x * blockDim.x + threadIdx.x) >> 5;
  const int NT   = Dn / 128;                 // 8
  const int m0   = (wid / NT) * 64;
  const int n0   = (wid % NT) * 128;
  const int ln   = lane & 15, hh = lane >> 4;

  v8f acc[4][8];
#pragma unroll
  for (int i = 0; i < 4; ++i)
#pragma unroll
    for (int j = 0; j < 8; ++j) acc[i][j] = vzero();

  for (int kc = 0; kc < Dn / 32; ++kc) {
    const int kbase = kc * 32;
    Frag a[4];
#pragma unroll
    for (int ms = 0; ms < 4; ++ms) {
      const unsigned short* ap = Ob + (size_t)(m0 + ms * 16 + ln) * Dn + kbase;
      __builtin_prefetch(ap + 128, 0, 0);
#pragma unroll
      for (int j = 0; j < 8; ++j)
        a[ms].u[j] = *(const unsigned int*)(ap + koffA(j, hh));
    }
#pragma unroll
    for (int ns = 0; ns < 8; ++ns) {
      Frag bf;
      const unsigned short* bp =
          Wt + (size_t)(n0 + ns * 16 + ln) * Dn + kbase + hh * 16;
      __builtin_prefetch(bp + 128, 0, 0);
#pragma unroll
      for (int j = 0; j < 8; ++j) bf.u[j] = *(const unsigned int*)(bp + 2 * j);
#pragma unroll
      for (int ms = 0; ms < 4; ++ms)
        acc[ms][ns] = __builtin_amdgcn_wmma_f32_16x16x32_bf16(
            false, a[ms].v, false, bf.v, (short)0, acc[ms][ns], false, false);
    }
  }

#pragma unroll
  for (int ms = 0; ms < 4; ++ms)
#pragma unroll
    for (int ns = 0; ns < 8; ++ns)
#pragma unroll
      for (int r = 0; r < 8; ++r)
        out[(size_t)(m0 + ms * 16 + r + 8 * hh) * Dn + n0 + ns * 16 + ln] =
            acc[ms][ns][r];
}

// ---------------------------------------------------------------------------
extern "C" void kernel_launch(void* const* d_in, const int* in_sizes, int n_in,
                              void* d_out, int out_size, void* d_ws, size_t ws_size,
                              hipStream_t stream) {
  (void)in_sizes; (void)n_in; (void)out_size; (void)ws_size;
  const float* x     = (const float*)d_in[0];
  const float* Wqkv  = (const float*)d_in[1];
  const float* Wproj = (const float*)d_in[2];
  const float* Wrel  = (const float*)d_in[3];

  unsigned short* ws = (unsigned short*)d_ws;
  const size_t per = (size_t)Bn * Hn * Tn * HDn;   // 8M bf16 elements
  unsigned short* Qb    = ws;                       //  [B,H,T,64]
  unsigned short* Kb    = ws + per;                 //  [B,H,T,64]
  unsigned short* Vt    = ws + 2 * per;             //  [B,H,64,T]
  unsigned short* Ob    = ws + 3 * per;             //  [B,T,D]
  unsigned short* Xb    = ws + 4 * per;             //  [8192,1024]
  unsigned short* Wqt   = ws + 5 * per;             //  [3072,1024]
  unsigned short* Wpt   = Wqt + (size_t)NQKV * Dn;  //  [1024,1024]

  // one-time conversions (hoists all f32->bf16 VALU out of GEMM loops)
  cvt_x_kernel<<<(MR * Dn / 4 + 255) / 256, 256, 0, stream>>>(x, Xb, MR * Dn / 4);
  cvt_t_kernel<<<(Dn * NQKV + 255) / 256, 256, 0, stream>>>(Wqkv, Wqt, Dn, NQKV);
  cvt_t_kernel<<<(Dn * Dn + 255) / 256, 256, 0, stream>>>(Wproj, Wpt, Dn, Dn);

  // QKV GEMM: (8192/64)*(3072/128) = 3072 waves -> 768 blocks of 4 waves
  qkv_gemm_kernel<<<768, 128, 0, stream>>>(Xb, Wqt, Qb, Kb, Vt);
  // Attention: B*H*(T/32) = 4096 waves -> 1024 blocks
  attn_kernel<<<1024, 128, 0, stream>>>(Qb, Kb, Vt, Wrel, Ob);
  // Proj GEMM: (8192/64)*(1024/128) = 1024 waves -> 256 blocks
  proj_gemm_kernel<<<256, 128, 0, stream>>>(Ob, Wpt, (float*)d_out);
}
